// CausalConvSelfAttention_13357348290662
// MI455X (gfx1250) — matrix operand order repaired and checked by reading, not compiled, not run on hardware
//
#include <hip/hip_runtime.h>

typedef __bf16 bf16_t;
typedef __attribute__((ext_vector_type(16))) __bf16 v16bf;
typedef __attribute__((ext_vector_type(8)))  __bf16 v8bf;
typedef __attribute__((ext_vector_type(4)))  __bf16 v4bf;
typedef __attribute__((ext_vector_type(8)))  float  v8f;
typedef __attribute__((ext_vector_type(4)))  float  v4f;

#define BSZ 2
#define T   4096
#define D   1024
#define KW  4

static constexpr int NX = BSZ * T * D;   // 8388608 X elements
static constexpr int NW = KW * D * D;    // 4194304 W elements

// ---------------------------------------------------------------------------
// Pass 1: split fp32 -> (hi, lo) bf16 pair.  x = hi + lo exactly captures the
// top ~16 mantissa bits; the 3-product WMMA scheme then gives ~fp32 accuracy.
// ---------------------------------------------------------------------------
__global__ __launch_bounds__(256) void split_bf16_kernel(
    const float* __restrict__ src, bf16_t* __restrict__ hi,
    bf16_t* __restrict__ lo, int n4)
{
    int i = blockIdx.x * blockDim.x + threadIdx.x;
    if (i >= n4) return;
    v4f x = ((const v4f*)src)[i];
    v4bf h, l;
#pragma unroll
    for (int e = 0; e < 4; ++e) {
        float xe = x[e];
        __bf16 hh = (__bf16)xe;          // RNE convert
        h[e] = hh;
        l[e] = (__bf16)(xe - (float)hh); // residual is exact in f32
    }
    ((v4bf*)hi)[i] = h;
    ((v4bf*)lo)[i] = l;
}

static __device__ __forceinline__ v16bf cat16(v8bf a, v8bf b) {
    return __builtin_shufflevector(a, b, 0, 1, 2, 3, 4, 5, 6, 7,
                                         8, 9, 10, 11, 12, 13, 14, 15);
}

// ---------------------------------------------------------------------------
// Pass 2: WMMA GEMM.  One wave -> 16(t) x 64(e) output tile, K = 4*1024 with
// the 4 causal shifts folded into the K loop.  EDGE=true only for t0==0.
// ---------------------------------------------------------------------------
template <bool EDGE>
__device__ __forceinline__ void tile_compute(
    const bf16_t* __restrict__ Xh, const bf16_t* __restrict__ Xl,
    const bf16_t* __restrict__ Wh, const bf16_t* __restrict__ Wl,
    float* __restrict__ out, int b, int t0, int e0, int lane)
{
    const int m   = lane & 15;   // A row / B col / C col within tile
    const int hs  = lane >> 4;   // which 16-lane half
    const int klo = hs * 16;     // A k sub-offset inside a 32-K block
    const int kof = hs * 16;     // B k sub-offset inside a 32-K block
    const v8bf z8 = {};

    v8f c[4] = {v8f{}, v8f{}, v8f{}, v8f{}};

    for (int i = 0; i < KW; ++i) {
        int row = t0 + m - i;
        bool valid = true;
        if (EDGE) { valid = (row >= 0); if (!valid) row = 0; }

        const bf16_t* pah = Xh + (size_t)(b * T + row) * D + klo;
        const bf16_t* pal = Xl + (size_t)(b * T + row) * D + klo;

        const bf16_t* pbh[4];
        const bf16_t* pbl[4];
#pragma unroll
        for (int j = 0; j < 4; ++j) {
            size_t wr = (size_t)(i * D + e0 + j * 16 + m);
            pbh[j] = Wh + wr * D + kof;
            pbl[j] = Wl + wr * D + kof;
        }

        for (int kb = 0; kb < D; kb += 32) {
            // A fragment (16x32 bf16): lanes 0-15 hold K 0-15,
            // lanes 16-31 hold K 16-31 (same order as B).
            v8bf ah0 = *(const v8bf*)(pah + kb);
            v8bf ah1 = *(const v8bf*)(pah + kb + 8);
            v8bf al0 = *(const v8bf*)(pal + kb);
            v8bf al1 = *(const v8bf*)(pal + kb + 8);
            if (EDGE) {
                ah0 = valid ? ah0 : z8;  ah1 = valid ? ah1 : z8;
                al0 = valid ? al0 : z8;  al1 = valid ? al1 : z8;
            }
            v16bf Ah = cat16(ah0, ah1);
            v16bf Al = cat16(al0, al1);

#pragma unroll
            for (int j = 0; j < 4; ++j) {
                // B fragment (32x16): lane = column N, 16 consecutive K per
                // lane half.
                v16bf Bh = cat16(*(const v8bf*)(pbh[j] + kb),
                                 *(const v8bf*)(pbh[j] + kb + 8));
                v16bf Bl = cat16(*(const v8bf*)(pbl[j] + kb),
                                 *(const v8bf*)(pbl[j] + kb + 8));
                c[j] = __builtin_amdgcn_wmma_f32_16x16x32_bf16(
                        false, Ah, false, Bh, (short)0, c[j], false, false);
                c[j] = __builtin_amdgcn_wmma_f32_16x16x32_bf16(
                        false, Ah, false, Bl, (short)0, c[j], false, false);
                c[j] = __builtin_amdgcn_wmma_f32_16x16x32_bf16(
                        false, Al, false, Bh, (short)0, c[j], false, false);
            }
        }
    }

    // C/D layout: VGPR r, lanes 0-15 -> M=r, lanes 16-31 -> M=r+8; N=lane&15.
#pragma unroll
    for (int j = 0; j < 4; ++j) {
#pragma unroll
        for (int r = 0; r < 8; ++r) {
            int trow = t0 + r + 8 * hs;
            int col  = e0 + 16 * j + m;
            out[(size_t)(b * T + trow) * D + col] = c[j][r];
        }
    }
}

__global__ __launch_bounds__(256) void ccsa_wmma_kernel(
    const bf16_t* __restrict__ Xh, const bf16_t* __restrict__ Xl,
    const bf16_t* __restrict__ Wh, const bf16_t* __restrict__ Wl,
    float* __restrict__ out)
{
    int lane = threadIdx.x & 31;
    int wid  = threadIdx.x >> 5;
    int w    = blockIdx.x * 8 + wid;        // 8192 waves total
    int e0   = (w & 15) * 64;               // 16 e-tiles of width 64
    int t0   = ((w >> 4) & 255) * 16;       // 256 t-tiles of height 16
    int b    = w >> 12;                     // 2 batches

    if (t0 == 0) tile_compute<true >(Xh, Xl, Wh, Wl, out, b, t0, e0, lane);
    else         tile_compute<false>(Xh, Xl, Wh, Wl, out, b, t0, e0, lane);
}

extern "C" void kernel_launch(void* const* d_in, const int* in_sizes, int n_in,
                              void* d_out, int out_size, void* d_ws, size_t ws_size,
                              hipStream_t stream)
{
    const float* X = (const float*)d_in[0];
    const float* W = (const float*)d_in[1];
    float* out = (float*)d_out;

    bf16_t* Xh = (bf16_t*)d_ws;
    bf16_t* Xl = Xh + NX;
    bf16_t* Wh = Xl + NX;
    bf16_t* Wl = Wh + NW;   // total ~50.3 MB of scratch

    split_bf16_kernel<<<(NX / 4 + 255) / 256, 256, 0, stream>>>(X, Xh, Xl, NX / 4);
    split_bf16_kernel<<<(NW / 4 + 255) / 256, 256, 0, stream>>>(W, Wh, Wl, NW / 4);

    // 2 * 256 * 16 = 8192 wave-tiles, 8 waves per block -> 1024 blocks
    ccsa_wmma_kernel<<<1024, 256, 0, stream>>>(Xh, Xl, Wh, Wl, out);
}
